// CrossAttention_65146063946571
// MI455X (gfx1250) — matrix operand order repaired and checked by reading, not compiled, not run on hardware
//
#include <hip/hip_runtime.h>

// ---------------- constants ----------------
#define NTOK   49          // tokens per window
#define NPAD   64          // padded to 4 x 16 tiles
#define DIMV   96
#define HEADS  3
#define HD     32
#define NWIN   1024
#define BWIN   16384
#define SCALE_F 0.17677669529663687f   // 32^-0.5

typedef __attribute__((ext_vector_type(16))) _Float16 v16h;
typedef __attribute__((ext_vector_type(8)))  _Float16 v8h;
typedef __attribute__((ext_vector_type(4)))  _Float16 v4h;
typedef __attribute__((ext_vector_type(8)))  float    v8f;

union F16x16 { v16h v; v8h h[2]; };

// A fragment (16x32 f16, MxK): lane L (g=L>>4, r=L&15) row M=r.
// halves 0..7  : K = kbase + 8g + 0..7
// halves 8..15 : K = kbase + 16 + 8g + 0..7
__device__ __forceinline__ v16h fragA(const _Float16* p0 /* at (row0,kbase) */,
                                      int stride, int lane) {
  int g = lane >> 4, r = lane & 15;
  const _Float16* p = p0 + r * stride + 8 * g;
  F16x16 f;
  f.h[0] = *(const v8h*)(p);
  f.h[1] = *(const v8h*)(p + 16);
  return f.v;
}

// B fragment (32x16 f16, KxN), stored as [N][K] row-major:
// lane L (g=L>>4, c=L&15) column N=c, halves 0..7: K = kbase + 8g + 0..7, halves 8..15: K = kbase + 16 + 8g + 0..7
__device__ __forceinline__ v16h fragB(const _Float16* p0 /* at (n0,kbase) */,
                                      int stride, int lane) {
  int g = lane >> 4, c = lane & 15;
  const _Float16* p = p0 + c * stride + 8 * g;
  F16x16 f;
  f.h[0] = *(const v8h*)(p);
  f.h[1] = *(const v8h*)(p + 16);
  return f.v;
}

#define WMMA_F16(A, B, C) \
  __builtin_amdgcn_wmma_f32_16x16x32_f16(false, (A), false, (B), (short)0, (C), false, false)

// ---------------- prologue: weight f32->f16 + bias gather ----------------
__global__ void prep_kernel(const float* __restrict__ Wqkv_d,
                            const float* __restrict__ Wqkv_r,
                            const float* __restrict__ btab,
                            const float* __restrict__ Wproj,
                            const int*   __restrict__ relidx,
                            _Float16* __restrict__ wq,
                            _Float16* __restrict__ wkv,
                            _Float16* __restrict__ wp,
                            float* __restrict__ bias3) {
  int i0 = blockIdx.x * blockDim.x + threadIdx.x;
  int stp = gridDim.x * blockDim.x;
  for (int t = i0; t < 96 * 96; t += stp)  wq[t]  = (_Float16)Wqkv_r[t];            // q rows 0..95
  for (int t = i0; t < 192 * 96; t += stp) wkv[t] = (_Float16)Wqkv_d[96 * 96 + t];  // k,v rows 96..287
  for (int t = i0; t < 96 * 96; t += stp)  wp[t]  = (_Float16)Wproj[t];
  for (int t = i0; t < HEADS * NTOK * NTOK; t += stp) {
    int h = t / (NTOK * NTOK), r = t % (NTOK * NTOK);
    bias3[t] = btab[relidx[r] * HEADS + h];
  }
}

// ---------------- main fused attention kernel ----------------
__global__ __launch_bounds__(256)
void CrossAttention_65146063946571_kernel(const float* __restrict__ Xd_all,
                                          const float* __restrict__ Xr_all,
                                          const float* __restrict__ mask,
                                          const float* __restrict__ bq_r,   // bqkv_rescaler
                                          const float* __restrict__ bq_d,   // bqkv_rescaled
                                          const float* __restrict__ bproj,
                                          const _Float16* __restrict__ wq,
                                          const _Float16* __restrict__ wkv,
                                          const _Float16* __restrict__ wp,
                                          const float* __restrict__ bias3,
                                          float* __restrict__ out) {
  // 60 KB LDS arena with stage-based aliasing
  __shared__ __align__(16) char smem[61440];
  _Float16* sXd = (_Float16*)(smem);           // [64][96] f16, stages 1-2
  _Float16* sXr = (_Float16*)(smem + 12288);   // [64][96] f16, stages 1-2
  _Float16* sQ  = (_Float16*)(smem + 24576);   // [3][64][32] scaled q  ([token][feat])
  _Float16* sK  = (_Float16*)(smem + 36864);   // [3][64][32]           ([token][feat])
  _Float16* sVT = (_Float16*)(smem + 49152);   // [3][32][64] v transposed ([feat][token])
  _Float16* sP  = (_Float16*)(smem);           // [3][64][64] ([query][key]), aliases sXd/sXr
  _Float16* sAO = (_Float16*)(smem + 24576);   // [64][96]   ([token][feat]), aliases sQ
  float*    sOut= (float*)(smem);              // [64][96] f32 (fits 24576B), aliases sP

  const int b    = blockIdx.x;
  const int tid  = threadIdx.x;
  const int wid  = __builtin_amdgcn_readfirstlane(tid >> 5);
  const int lane = tid & 31;
  const int g    = lane >> 4;
  const int cl   = lane & 15;

  __builtin_prefetch(wq, 0, 0);
  __builtin_prefetch(wkv, 0, 0);

  // ---- stage 1: global f32 -> LDS f16 (rows 49..63 zero padded) ----
  {
    const float4* srcd = (const float4*)(Xd_all + (size_t)b * NTOK * DIMV);
    const float4* srcr = (const float4*)(Xr_all + (size_t)b * NTOK * DIMV);
    for (int i = tid; i < (NTOK * DIMV) / 4; i += 256) {   // 1176 vec4 per matrix
      float4 td = srcd[i];
      float4 tr = srcr[i];
      int f = i * 4, row = f / DIMV, col = f - row * DIMV;
      v4h hd, hr;
      hd[0] = (_Float16)td.x; hd[1] = (_Float16)td.y; hd[2] = (_Float16)td.z; hd[3] = (_Float16)td.w;
      hr[0] = (_Float16)tr.x; hr[1] = (_Float16)tr.y; hr[2] = (_Float16)tr.z; hr[3] = (_Float16)tr.w;
      *(v4h*)(sXd + row * DIMV + col) = hd;
      *(v4h*)(sXr + row * DIMV + col) = hr;
    }
    for (int i = tid; i < (NPAD - NTOK) * DIMV; i += 256) {
      sXd[NTOK * DIMV + i] = (_Float16)0.0f;
      sXr[NTOK * DIMV + i] = (_Float16)0.0f;
    }
  }
  __syncthreads();

  // ---- stage 2: QKV projections via WMMA ----
  // q and k computed transposed: C = W * X^T  (A = weight rows = 6 Mtiles of features,
  // B = X row-major = 4 Ntiles of tokens).  Lane's 8 C values are 8 consecutive
  // features of one token -> single packed ds_store_b128 into [token][feat] layout.
  for (int job = wid; job < 48; job += 8) {
    int isk = job >= 24;                 // wave-uniform (scalar)
    int t = job & 23 ? job % 24 : job % 24;  // job%24
    t = job % 24;
    int mt = t / 4, nt = t % 4;          // mt: feature tile(6), nt: token tile(4)
    const _Float16* W  = isk ? wkv : wq;
    const _Float16* B  = isk ? sXd : sXr;
    const float*    bi = isk ? (bq_d + 96) : bq_r;
    _Float16*       dst= isk ? sK : sQ;
    const float     sc = isk ? 1.0f : SCALE_F;
    v8f acc = {};
#pragma unroll
    for (int kt = 0; kt < 3; ++kt) {
      v16h fa = fragA(W + mt * 16 * DIMV + kt * 32, DIMV, lane);
      v16h fb = fragB(B + nt * 16 * DIMV + kt * 32, DIMV, lane);
      acc = WMMA_F16(fa, fb, acc);
    }
    int m  = nt * 16 + cl;        // token (C column)
    int nb = mt * 16 + 8 * g;     // feature row base; j adds 0..7 (stays in one head)
    int h = nb >> 5, db = nb & 31;
    const float4* b4 = (const float4*)(bi + nb);
    float4 b0 = b4[0], b1 = b4[1];
    v8h pk;
    pk[0] = (_Float16)((acc[0] + b0.x) * sc);
    pk[1] = (_Float16)((acc[1] + b0.y) * sc);
    pk[2] = (_Float16)((acc[2] + b0.z) * sc);
    pk[3] = (_Float16)((acc[3] + b0.w) * sc);
    pk[4] = (_Float16)((acc[4] + b1.x) * sc);
    pk[5] = (_Float16)((acc[5] + b1.y) * sc);
    pk[6] = (_Float16)((acc[6] + b1.z) * sc);
    pk[7] = (_Float16)((acc[7] + b1.w) * sc);
    *(v8h*)(dst + h * 2048 + m * HD + db) = pk;
  }
  // v in natural orientation: C = X * W^T; lane's 8 values = 8 consecutive tokens of
  // one feature -> single packed store into transposed sVT[feat][token] layout.
  for (int job = wid; job < 24; job += 8) {
    int mt = job / 6, nt = job % 6;      // mt: token tile(4), nt: feature tile(6)
    v8f acc = {};
#pragma unroll
    for (int kt = 0; kt < 3; ++kt) {
      v16h fa = fragA(sXd + mt * 16 * DIMV + kt * 32, DIMV, lane);
      v16h fb = fragB(wkv + 96 * 96 + nt * 16 * DIMV + kt * 32, DIMV, lane);
      acc = WMMA_F16(fa, fb, acc);
    }
    int n = nt * 16 + cl;                // feature (C column)
    int h = n >> 5, d = n & 31;
    int mb = mt * 16 + 8 * g;            // token row base
    float bv = bq_d[192 + n];
    v8h pk;
#pragma unroll
    for (int j = 0; j < 8; ++j) pk[j] = (_Float16)(acc[j] + bv);
    *(v8h*)(sVT + h * 2048 + d * NPAD + mb) = pk;
  }
  __syncthreads();

  // ---- stage 3: fused S = q k^T (+bias+mask) + row softmax -> P (f16) ----
  // 12 jobs: (head, query Mtile); wave owns 16 query rows for one head.
  for (int job = wid; job < 12; job += 8) {
    int h = job / 4, mt = job % 4;
    v16h fq = fragA(sQ + h * 2048 + mt * 16 * HD, HD, lane);
    v8f accS[4];
#pragma unroll
    for (int nt = 0; nt < 4; ++nt) {
      v8f z = {};
      v16h fk = fragB(sK + h * 2048 + nt * 16 * HD, HD, lane);
      accS[nt] = WMMA_F16(fq, fk, z);
    }
    const float* brow  = bias3 + h * (NTOK * NTOK);
    const float* maskp = mask + (size_t)(b & (NWIN - 1)) * (NTOK * NTOK);
#pragma unroll
    for (int j = 0; j < 8; ++j) {
      int m = mt * 16 + j + 8 * g;           // query row
      bool rv = (m < NTOK);
      int mc = rv ? m : (NTOK - 1);
      float xx[4];
      bool  cv[4];
#pragma unroll
      for (int nt = 0; nt < 4; ++nt) {
        int c = nt * 16 + cl;                // key column
        cv[nt] = rv && (c < NTOK);
        int cc = (c < NTOK) ? c : (NTOK - 1);
        int idx = mc * NTOK + cc;            // clamped: loads always valid, no branch
        float add = brow[idx] + maskp[idx];
        xx[nt] = cv[nt] ? (accS[nt][j] + add) : -3.0e38f;
      }
      float mx = fmaxf(fmaxf(xx[0], xx[1]), fmaxf(xx[2], xx[3]));
#pragma unroll
      for (int o = 8; o >= 1; o >>= 1) mx = fmaxf(mx, __shfl_xor(mx, o, 32));
      float e[4], s = 0.0f;
#pragma unroll
      for (int nt = 0; nt < 4; ++nt) {
        e[nt] = cv[nt] ? __expf(xx[nt] - mx) : 0.0f;
        s += e[nt];
      }
#pragma unroll
      for (int o = 8; o >= 1; o >>= 1) s += __shfl_xor(s, o, 32);
      float inv = rv ? (1.0f / s) : 0.0f;
#pragma unroll
      for (int nt = 0; nt < 4; ++nt) {
        int c = nt * 16 + cl;
        sP[h * 4096 + m * NPAD + c] = (_Float16)(e[nt] * inv);
      }
    }
  }
  __syncthreads();

  // ---- stage 4: O^T = V^T * P^T via WMMA (both operands already in layout) ----
  // 24 jobs: (head, 2 feature tiles, 4 token tiles); packed store into sAO[token][feat].
  for (int job = wid; job < 24; job += 8) {
    int h = job / 8, t = job % 8, dt = t / 4, tt = t % 4;
    v8f acc = {};
#pragma unroll
    for (int kt = 0; kt < 2; ++kt) {      // key dim 64 -> 2 steps
      v16h fa = fragA(sVT + h * 2048 + dt * 16 * NPAD + kt * 32, NPAD, lane);
      v16h fb = fragB(sP  + h * 4096 + tt * 16 * NPAD + kt * 32, NPAD, lane);
      acc = WMMA_F16(fa, fb, acc);
    }
    int m  = tt * 16 + cl;                // token (C column)
    int dbse = dt * 16 + 8 * g;           // feature row base
    v8h pk;
#pragma unroll
    for (int j = 0; j < 8; ++j) pk[j] = (_Float16)acc[j];
    *(v8h*)(sAO + m * DIMV + h * HD + dbse) = pk;
  }
  __syncthreads();

  // ---- stage 5: OUT^T = Wp * AO^T, f32 rows staged in LDS ----
  for (int job = wid; job < 24; job += 8) {
    int mt = job / 4, tt = job % 4;       // mt: out-feature tile(6), tt: token tile(4)
    v8f acc = {};
#pragma unroll
    for (int kt = 0; kt < 3; ++kt) {
      v16h fa = fragA(wp  + mt * 16 * DIMV + kt * 32, DIMV, lane);
      v16h fb = fragB(sAO + tt * 16 * DIMV + kt * 32, DIMV, lane);
      acc = WMMA_F16(fa, fb, acc);
    }
    int m  = tt * 16 + cl;                // token
    int nb = mt * 16 + 8 * g;             // out-feature base
    const float4* b4 = (const float4*)(bproj + nb);
    float4 b0 = b4[0], b1 = b4[1];
    float* dst = sOut + m * DIMV + nb;
    float4 o0, o1;
    o0.x = acc[0] + b0.x; o0.y = acc[1] + b0.y; o0.z = acc[2] + b0.z; o0.w = acc[3] + b0.w;
    o1.x = acc[4] + b1.x; o1.y = acc[5] + b1.y; o1.z = acc[6] + b1.z; o1.w = acc[7] + b1.w;
    *(float4*)(dst)     = o0;
    *(float4*)(dst + 4) = o1;             // rows >= 49 land in dead LDS slack, never stored
  }
  __syncthreads();

  // ---- stage 6: fully coalesced f32 output store ----
  {
    const float4* s4 = (const float4*)sOut;
    float4* o4 = (float4*)(out + (size_t)b * NTOK * DIMV);
    for (int i = tid; i < (NTOK * DIMV) / 4; i += 256) o4[i] = s4[i];
  }
}

// ---------------- host launcher ----------------
extern "C" void kernel_launch(void* const* d_in, const int* in_sizes, int n_in,
                              void* d_out, int out_size, void* d_ws, size_t ws_size,
                              hipStream_t stream) {
  (void)in_sizes; (void)n_in; (void)out_size; (void)ws_size;
  const float* rescaled = (const float*)d_in[0];
  const float* rescaler = (const float*)d_in[1];
  const float* mask     = (const float*)d_in[2];
  const float* Wqkv_d   = (const float*)d_in[3];
  const float* bqkv_d   = (const float*)d_in[4];
  const float* Wqkv_r   = (const float*)d_in[5];
  const float* bqkv_r   = (const float*)d_in[6];
  const float* btab     = (const float*)d_in[7];
  const float* Wproj    = (const float*)d_in[8];
  const float* bprojp   = (const float*)d_in[9];
  const int*   relidx   = (const int*)d_in[10];
  float*       outp     = (float*)d_out;

  char* ws = (char*)d_ws;
  _Float16* wq    = (_Float16*)(ws);           //  96x96 f16 = 18432 B
  _Float16* wkv   = (_Float16*)(ws + 18432);   // 192x96 f16 = 36864 B
  _Float16* wp    = (_Float16*)(ws + 55296);   //  96x96 f16 = 18432 B
  float*    bias3 = (float*)   (ws + 73728);   // 3x49x49 f32 = 28812 B

  prep_kernel<<<64, 256, 0, stream>>>(Wqkv_d, Wqkv_r, btab, Wproj, relidx,
                                      wq, wkv, wp, bias3);
  CrossAttention_65146063946571_kernel<<<BWIN, 256, 0, stream>>>(
      rescaled, rescaler, mask, bqkv_r, bqkv_d, bprojp, wq, wkv, wp, bias3, outp);
}
